// RNNIBase_90915867722300
// MI455X (gfx1250) — matrix operand order repaired and checked by reading, not compiled, not run on hardware
//
#include <hip/hip_runtime.h>
#include <stdint.h>

// Problem dims (fixed by the reference)
#define Bb   128
#define Tt   512
#define Ff   128
#define Hh   512
#define H3   1536
#define KIN  256   // cell1 input width = 2*F (x ++ mask)

// Persistent-kernel grid: must be co-resident (MI455X has far more WGP slots)
#define NWG      64
#define BLOCK    256
#define NWAVES   (NWG * (BLOCK / 32))
#define NTHREADS (NWG * BLOCK)

typedef __attribute__((ext_vector_type(16))) __bf16 v16bf;
typedef __attribute__((ext_vector_type(8)))  __bf16 v8bf;
typedef __attribute__((ext_vector_type(8)))  float  v8f;

// ---------- helpers ----------

__device__ __forceinline__ uint16_t f2bfu(float f) {
  uint32_t u = __float_as_uint(f);
  uint32_t r = ((u >> 16) & 1u) + 0x7fffu;   // round-to-nearest-even
  return (uint16_t)((u + r) >> 16);
}

__device__ __forceinline__ float sigm(float v) {
  return 1.0f / (1.0f + __expf(-v));
}

__device__ __forceinline__ float tanh_f(float v) {
  float c = fminf(fmaxf(v, -15.0f), 15.0f);
  float e = __expf(2.0f * c);
  return (e - 1.0f) / (e + 1.0f);
}

__device__ __forceinline__ v16bf ldfrag(const uint16_t* p0, const uint16_t* p1) {
  union { v16bf v; v8bf h[2]; } u;
  u.h[0] = *(const v8bf*)p0;
  u.h[1] = *(const v8bf*)p1;
  return u.v;
}

// A-matrix 16x32 bf16 fragment (ISA 7.12.2): lanes 0-15 hold M=lane,
// elems 0..7 = K kb+0..7, elems 8..15 = K kb+16..23; lanes 16-31 hold
// M=lane-16 with K offsets +8 / +24.
__device__ __forceinline__ v16bf load_a(const uint16_t* A, int lda,
                                        int m0, int kb, int lane) {
  const uint16_t* p = A + (size_t)(m0 + (lane & 15)) * lda + kb + ((lane >> 4) << 3);
  return ldfrag(p, p + 16);
}

// B-matrix 32x16 bf16 fragment: lane holds column N = n0 + (lane&15),
// elems 0..7 = K kb+0..7, elems 8..15 = K kb+16..23 (lanes 16-31: +8 / +24),
// matching load_a. Weights are stored [N_total, K] row-major (= B^T).
__device__ __forceinline__ v16bf load_b(const uint16_t* W, int ldw,
                                        int n0, int kb, int lane) {
  const uint16_t* p = W + (size_t)(n0 + (lane & 15)) * ldw + kb + ((lane >> 4) << 3);
  return ldfrag(p, p + 16);
}

__device__ __forceinline__ v8f wmma_bf16(v16bf a, v16bf b, v8f c) {
  return __builtin_amdgcn_wmma_f32_16x16x32_bf16(false, a, false, b,
                                                 (short)0, c, false, false);
}

// Sense-reversing device-wide barrier (persistent kernel, all WGs resident).
__device__ __forceinline__ void grid_barrier(unsigned* bar) {
  __syncthreads();
  if (threadIdx.x == 0) {
    __threadfence();
    unsigned gen = __hip_atomic_load(&bar[1], __ATOMIC_RELAXED, __HIP_MEMORY_SCOPE_AGENT);
    unsigned arrived = __hip_atomic_fetch_add(&bar[0], 1u, __ATOMIC_ACQ_REL,
                                              __HIP_MEMORY_SCOPE_AGENT);
    if (arrived == NWG - 1) {
      __hip_atomic_store(&bar[0], 0u, __ATOMIC_RELAXED, __HIP_MEMORY_SCOPE_AGENT);
      __hip_atomic_fetch_add(&bar[1], 1u, __ATOMIC_RELEASE, __HIP_MEMORY_SCOPE_AGENT);
    } else {
      while (__hip_atomic_load(&bar[1], __ATOMIC_ACQUIRE,
                               __HIP_MEMORY_SCOPE_AGENT) == gen) {
        __builtin_amdgcn_s_sleep(1);
      }
    }
  }
  __syncthreads();
}

// ---------- prologue: f32 -> bf16 weight conversion ----------

__global__ void cvt_bf16(const float* __restrict__ src,
                         uint16_t* __restrict__ dst, int n) {
  int i = blockIdx.x * blockDim.x + threadIdx.x;
  if (i < n) dst[i] = f2bfu(src[i]);
}

// ---------- the persistent RNN kernel ----------

__global__ __launch_bounds__(BLOCK) void rnni_persistent(
    const float* __restrict__ x, const uint8_t* __restrict__ mask,
    const float* __restrict__ bih1, const float* __restrict__ bhh1,
    const float* __restrict__ bih2, const float* __restrict__ bhh2,
    const float* __restrict__ bro,
    const uint16_t* __restrict__ wih1, const uint16_t* __restrict__ whh1,
    const uint16_t* __restrict__ wih2, const uint16_t* __restrict__ whh2,
    const uint16_t* __restrict__ wro,
    float* __restrict__ gh1, float* __restrict__ gh2,
    uint16_t* __restrict__ xin,
    float* __restrict__ h1, float* __restrict__ h2,
    uint16_t* __restrict__ h1b, uint16_t* __restrict__ h2b,
    float* __restrict__ xout, float* __restrict__ hout,
    float* __restrict__ h1out, float* __restrict__ h2out,
    unsigned* __restrict__ bar) {
  const int lane   = threadIdx.x & 31;
  const int wid    = blockIdx.x * (BLOCK >> 5) + (threadIdx.x >> 5);
  const int gtid   = blockIdx.x * BLOCK + threadIdx.x;
  const int colSel = lane & 15;
  const int rowHi  = (lane >> 4) << 3;   // +0 or +8 on the M index of C/D rows

  for (int t = 0; t < Tt; ++t) {
    // ---- record h_out[t] = h2 (pre-update): one float4 per thread ----
    {
      int i4 = gtid << 2;                       // NTHREADS*4 == Bb*Hh exactly
      int b = i4 >> 9, j = i4 & (Hh - 1);
      *(float4*)&hout[((size_t)b * Tt + t) * Hh + j] = *(const float4*)&h2[i4];
    }

    // ---- Phase A (400 jobs, one round over 512 waves):
    //   192 jobs: gh1 = h1*Whh1^T + bhh1  (16x64 strip each, A reused 4x)
    //   192 jobs: gh2 = h2*Whh2^T + bhh2
    //    16 jobs: x_hat = h2*Wro^T + bro fused with impute/concat/x_out ----
    for (int job = wid; job < 400; job += NWAVES) {
      if (job < 384) {
        const bool c1 = job < 192;
        int jj = c1 ? job : job - 192;
        const uint16_t* hb = c1 ? h1b : h2b;
        const uint16_t* W  = c1 ? whh1 : whh2;
        const float* bh    = c1 ? bhh1 : bhh2;
        float* gh          = c1 ? gh1 : gh2;
        int m0 = (jj / 24) << 4;
        int n0 = (jj % 24) << 6;                // 64-wide N strip
        v8f acc[4];
#pragma unroll
        for (int c = 0; c < 4; ++c) {
          float bv = bh[n0 + (c << 4) + colSel];
#pragma unroll
          for (int i = 0; i < 8; ++i) acc[c][i] = bv;
        }
#pragma unroll 2
        for (int kb = 0; kb < Hh; kb += 32) {
          v16bf a = load_a(hb, Hh, m0, kb, lane);
#pragma unroll
          for (int c = 0; c < 4; ++c)
            acc[c] = wmma_bf16(a, load_b(W, Hh, n0 + (c << 4), kb, lane), acc[c]);
        }
#pragma unroll
        for (int c = 0; c < 4; ++c) {
          int col = n0 + (c << 4) + colSel;
#pragma unroll
          for (int i = 0; i < 8; ++i)
            gh[(size_t)(m0 + i + rowHi) * H3 + col] = acc[c][i];
        }
      } else {
        int jj = job - 384;                     // 0..15
        int m0 = (jj >> 1) << 4;
        int n0 = (jj & 1) << 6;                 // 0 or 64
        v8f acc[4];
#pragma unroll
        for (int c = 0; c < 4; ++c) {
          float bv = bro[n0 + (c << 4) + colSel];
#pragma unroll
          for (int i = 0; i < 8; ++i) acc[c][i] = bv;
        }
#pragma unroll 2
        for (int kb = 0; kb < Hh; kb += 32) {
          v16bf a = load_a(h2b, Hh, m0, kb, lane);
#pragma unroll
          for (int c = 0; c < 4; ++c)
            acc[c] = wmma_bf16(a, load_b(wro, Hh, n0 + (c << 4), kb, lane), acc[c]);
        }
#pragma unroll
        for (int c = 0; c < 4; ++c) {
          int col = n0 + (c << 4) + colSel;
#pragma unroll
          for (int i = 0; i < 8; ++i) {
            int b = m0 + i + rowHi;
            size_t idx = ((size_t)b * Tt + t) * Ff + col;
            float xh = acc[c][i];
            xout[idx] = xh;
            uint8_t m = mask[idx];
            float xv = m ? x[idx] : xh;
            xin[(size_t)b * KIN + col]      = f2bfu(xv);
            xin[(size_t)b * KIN + Ff + col] = m ? (uint16_t)0x3F80u : (uint16_t)0;
          }
        }
      }
    }
    grid_barrier(bar);

    // ---- Phase B: cell1. Each wave owns one 16x16 (b,j) block and computes
    //      all 3 gate tiles of gi1 (K=256) then the GRU combine in-register. ----
    for (int job = wid; job < 256; job += NWAVES) {
      int m0 = (job >> 5) << 4;
      int j0 = (job & 31) << 4;
      int col = j0 + colSel;
      v8f ar, az, an;
      float vr = bih1[col], vz = bih1[Hh + col], vn = bih1[2 * Hh + col];
#pragma unroll
      for (int i = 0; i < 8; ++i) { ar[i] = vr; az[i] = vz; an[i] = vn; }
#pragma unroll 2
      for (int kb = 0; kb < KIN; kb += 32) {
        v16bf a = load_a(xin, KIN, m0, kb, lane);
        ar = wmma_bf16(a, load_b(wih1, KIN, j0,          kb, lane), ar);
        az = wmma_bf16(a, load_b(wih1, KIN, Hh + j0,     kb, lane), az);
        an = wmma_bf16(a, load_b(wih1, KIN, 2 * Hh + j0, kb, lane), an);
      }
#pragma unroll
      for (int i = 0; i < 8; ++i) {
        int b = m0 + i + rowHi;
        const float* g = gh1 + (size_t)b * H3;
        float r  = sigm(ar[i] + g[col]);
        float z  = sigm(az[i] + g[Hh + col]);
        float nn = tanh_f(an[i] + r * g[2 * Hh + col]);
        float hp = h1[(size_t)b * Hh + col];
        float hv = (1.0f - z) * nn + z * hp;
        h1[(size_t)b * Hh + col]  = hv;
        h1b[(size_t)b * Hh + col] = f2bfu(hv);
      }
    }
    grid_barrier(bar);

    // ---- Phase C: cell2, identical shape but K=512 with A = h1 (bf16). ----
    for (int job = wid; job < 256; job += NWAVES) {
      int m0 = (job >> 5) << 4;
      int j0 = (job & 31) << 4;
      int col = j0 + colSel;
      v8f ar, az, an;
      float vr = bih2[col], vz = bih2[Hh + col], vn = bih2[2 * Hh + col];
#pragma unroll
      for (int i = 0; i < 8; ++i) { ar[i] = vr; az[i] = vz; an[i] = vn; }
#pragma unroll 2
      for (int kb = 0; kb < Hh; kb += 32) {
        v16bf a = load_a(h1b, Hh, m0, kb, lane);
        ar = wmma_bf16(a, load_b(wih2, Hh, j0,          kb, lane), ar);
        az = wmma_bf16(a, load_b(wih2, Hh, Hh + j0,     kb, lane), az);
        an = wmma_bf16(a, load_b(wih2, Hh, 2 * Hh + j0, kb, lane), an);
      }
#pragma unroll
      for (int i = 0; i < 8; ++i) {
        int b = m0 + i + rowHi;
        const float* g = gh2 + (size_t)b * H3;
        float r  = sigm(ar[i] + g[col]);
        float z  = sigm(az[i] + g[Hh + col]);
        float nn = tanh_f(an[i] + r * g[2 * Hh + col]);
        float hp = h2[(size_t)b * Hh + col];
        float hv = (1.0f - z) * nn + z * hp;
        h2[(size_t)b * Hh + col]  = hv;
        h2b[(size_t)b * Hh + col] = f2bfu(hv);
      }
    }
    grid_barrier(bar);
  }

  // ---- final states: one float4 per thread per array ----
  {
    int i4 = gtid << 2;
    *(float4*)&h1out[i4] = *(const float4*)&h1[i4];
    *(float4*)&h2out[i4] = *(const float4*)&h2[i4];
  }
}

// ---------- host side ----------

extern "C" void kernel_launch(void* const* d_in, const int* in_sizes, int n_in,
                              void* d_out, int out_size, void* d_ws, size_t ws_size,
                              hipStream_t stream) {
  (void)in_sizes; (void)n_in; (void)out_size; (void)ws_size;
  const float*   xp    = (const float*)d_in[0];
  const uint8_t* maskp = (const uint8_t*)d_in[1];   // jax bool_ = 1 byte/elem
  const float*   Wih1  = (const float*)d_in[2];
  const float*   Whh1  = (const float*)d_in[3];
  const float*   bih1  = (const float*)d_in[4];
  const float*   bhh1  = (const float*)d_in[5];
  const float*   Wih2  = (const float*)d_in[6];
  const float*   Whh2  = (const float*)d_in[7];
  const float*   bih2  = (const float*)d_in[8];
  const float*   bhh2  = (const float*)d_in[9];
  const float*   Wro   = (const float*)d_in[10];
  const float*   bro   = (const float*)d_in[11];

  // workspace carve-up (256B-aligned regions)
  char* ws = (char*)d_ws;
  size_t off = 0;
  auto carve = [&](size_t bytes) -> void* {
    void* p = ws + off;
    off = (off + bytes + 255) & ~(size_t)255;
    return p;
  };
  unsigned* bar  = (unsigned*)carve(256);
  uint16_t* w1b  = (uint16_t*)carve((size_t)H3 * KIN * 2);  // Wih1 bf16
  uint16_t* wh1b = (uint16_t*)carve((size_t)H3 * Hh * 2);   // Whh1 bf16
  uint16_t* w2b  = (uint16_t*)carve((size_t)H3 * Hh * 2);   // Wih2 bf16
  uint16_t* wh2b = (uint16_t*)carve((size_t)H3 * Hh * 2);   // Whh2 bf16
  uint16_t* wrob = (uint16_t*)carve((size_t)Ff * Hh * 2);   // Wro  bf16
  float*    gh1  = (float*)carve((size_t)Bb * H3 * 4);
  float*    gh2  = (float*)carve((size_t)Bb * H3 * 4);
  uint16_t* xin  = (uint16_t*)carve((size_t)Bb * KIN * 2);
  float*    h1   = (float*)carve((size_t)Bb * Hh * 4);
  float*    h2   = (float*)carve((size_t)Bb * Hh * 4);
  uint16_t* h1b  = (uint16_t*)carve((size_t)Bb * Hh * 2);
  uint16_t* h2b  = (uint16_t*)carve((size_t)Bb * Hh * 2);

  // outputs: x_out [B,T,F], h_out [B,T,H], h1 [B,H], h2 [B,H] (flat, in order)
  float* out   = (float*)d_out;
  float* xout  = out;
  float* hout  = out + (size_t)Bb * Tt * Ff;
  float* h1out = hout + (size_t)Bb * Tt * Hh;
  float* h2out = h1out + (size_t)Bb * Hh;

  // deterministic per-call init (graph-capture safe)
  hipMemsetAsync(bar, 0, 256, stream);
  hipMemsetAsync(h1,  0, (size_t)Bb * Hh * 4, stream);
  hipMemsetAsync(h2,  0, (size_t)Bb * Hh * 4, stream);
  hipMemsetAsync(h1b, 0, (size_t)Bb * Hh * 2, stream);
  hipMemsetAsync(h2b, 0, (size_t)Bb * Hh * 2, stream);

  // weight conversion f32 -> bf16 (weights then live in L2 for the whole run)
  auto cvt = [&](const float* s, uint16_t* d, int n) {
    cvt_bf16<<<(n + 255) / 256, 256, 0, stream>>>(s, d, n);
  };
  cvt(Wih1, w1b,  H3 * KIN);
  cvt(Whh1, wh1b, H3 * Hh);
  cvt(Wih2, w2b,  H3 * Hh);
  cvt(Whh2, wh2b, H3 * Hh);
  cvt(Wro,  wrob, Ff * Hh);

  rnni_persistent<<<NWG, BLOCK, 0, stream>>>(
      xp, maskp, bih1, bhh1, bih2, bhh2, bro,
      w1b, wh1b, w2b, wh2b, wrob,
      gh1, gh2, xin, h1, h2, h1b, h2b,
      xout, hout, h1out, h2out, bar);
}
